// DERFAttention_12206297055278
// MI455X (gfx1250) — matrix operand order repaired and checked by reading, not compiled, not run on hardware
//
#include <hip/hip_runtime.h>
#include <hip/hip_bf16.h>
#include <math.h>

// ---------------------------------------------------------------------------
// DERF attention forward for MI455X (gfx1250, wave32, WMMA).
// Shapes fixed by the reference: B=4, S=1024, E=1024, H=16, DK=64, BH=64.
// ---------------------------------------------------------------------------

#define S_LEN 1024
#define E_DIM 1024
#define H_NUM 16
#define DKV   64
#define B_NUM 4
#define BH_NUM 64

typedef __attribute__((ext_vector_type(16))) _Float16 v16h;
typedef __attribute__((ext_vector_type(8)))  _Float16 v8h;
typedef __attribute__((ext_vector_type(8)))  float    v8f;

static __device__ inline v8f wmma_f16(v16h a, v16h b, v8f c) {
  // D = A(16x32 f16) * B(32x16 f16) + C(16x16 f32)
  return __builtin_amdgcn_wmma_f32_16x16x32_f16(
      /*neg_a=*/false, a, /*neg_b=*/false, b,
      /*c_mod=*/(short)0, c, /*reuse_a=*/false, /*reuse_b=*/false);
}

// Pack 8 floats (two float4) into 8 f16 for a 16-byte LDS store.
static __device__ inline v8h cvt_v8h(float4 a, float4 b) {
  v8h r;
  r[0] = (_Float16)a.x; r[1] = (_Float16)a.y; r[2] = (_Float16)a.z; r[3] = (_Float16)a.w;
  r[4] = (_Float16)b.x; r[5] = (_Float16)b.y; r[6] = (_Float16)b.z; r[7] = (_Float16)b.w;
  return r;
}

// A-fragment (16x32, f16) load from LDS, per ISA 7.12.2 16-bit A layout:
//  lanes 0-15 : M=lane,     K chunks {0..7, 16..23}
//  lanes16-31 : M=lane-16,  K chunks {8..15, 24..31}
static __device__ inline v16h load_a_frag(const _Float16* base, int ldm, int lane) {
  const int r    = lane & 15;
  const int koff = (lane >> 4) << 3;   // 0 or 8
  const _Float16* p = base + r * ldm;
  v8h lo = *(const v8h*)(p + koff);
  v8h hi = *(const v8h*)(p + 16 + koff);
  v16h out;
#pragma unroll
  for (int i = 0; i < 8; ++i) { out[i] = lo[i]; out[8 + i] = hi[i]; }
  return out;
}

// B-fragment (32x16, f16) load from LDS: lane l holds column n=l%16, K chunks {8h..8h+7, 16+8h..16+8h+7}.
static __device__ inline v16h load_b_frag(const _Float16* base, int ldm, int lane) {
  const int koff = (lane >> 4) << 3;   // 0 or 8
  const _Float16* p = base + (lane & 15);
  v16h out;
#pragma unroll
  for (int i = 0; i < 8; ++i) { out[i] = p[(koff + i) * ldm]; out[8 + i] = p[(16 + koff + i) * ldm]; }
  return out;
}

// ---------------------------------------------------------------------------
// K1: Y = (X @ W^T + bias) * scale ; X:[Mrows,K], W:[N,K].
// head_layout=1 -> write [bh][s][d] (bh=b*H+h), else row-major [Mrows,N].
// One 64x64 output tile per workgroup; 4 waves, each owns a 16-row stripe.
// ---------------------------------------------------------------------------
__global__ __launch_bounds__(128)
void gemm_xwt_kernel(const float* __restrict__ X, const float* __restrict__ W,
                     const float* __restrict__ bias, float* __restrict__ Y,
                     int Mrows, int N, int K, float scale, int head_layout) {
  __shared__ __align__(16) _Float16 At[64][32];   // A tile  (rows x K-chunk)
  __shared__ __align__(16) _Float16 Bt[32][64];   // Bt[k][n] = W[n][k]
  const int t = threadIdx.x;
  const int lane = t & 31;
  const int w = t >> 5;
  const int row0 = blockIdx.y * 64;
  const int col0 = blockIdx.x * 64;

  const v8f vzero = {0.f, 0.f, 0.f, 0.f, 0.f, 0.f, 0.f, 0.f};
  v8f acc[4];
#pragma unroll
  for (int c = 0; c < 4; ++c) acc[c] = vzero;

  const int sr = t >> 1;               // staging row / W row (0..63)
  const int sc = (t & 1) << 4;         // staging K-chunk (0 or 16)

  for (int k0 = 0; k0 < K; k0 += 32) {
    {
      // A tile: 128-bit global loads, 128-bit LDS stores
      const float4* xs = (const float4*)(X + (size_t)(row0 + sr) * K + k0 + sc);
      float4 f0 = xs[0], f1 = xs[1], f2 = xs[2], f3 = xs[3];
      *(v8h*)&At[sr][sc]     = cvt_v8h(f0, f1);
      *(v8h*)&At[sr][sc + 8] = cvt_v8h(f2, f3);

      // B tile: vector global loads, transposed scalar LDS stores
      const float4* ws4 = (const float4*)(W + (size_t)(col0 + sr) * K + k0 + sc);
      float4 g0 = ws4[0], g1 = ws4[1], g2 = ws4[2], g3 = ws4[3];
      float tmp[16] = {g0.x, g0.y, g0.z, g0.w, g1.x, g1.y, g1.z, g1.w,
                       g2.x, g2.y, g2.z, g2.w, g3.x, g3.y, g3.z, g3.w};
#pragma unroll
      for (int u = 0; u < 16; ++u) Bt[sc + u][sr] = (_Float16)tmp[u];
    }
    if (k0 + 32 < K) {                 // cover HBM latency for next K-step
      __builtin_prefetch(X + (size_t)(row0 + sr) * K + k0 + 32 + sc, 0, 1);
      __builtin_prefetch(W + (size_t)(col0 + sr) * K + k0 + 32 + sc, 0, 1);
    }
    __syncthreads();
    v16h a = load_a_frag(&At[w * 16][0], 32, lane);
#pragma unroll
    for (int c = 0; c < 4; ++c)
      acc[c] = wmma_f16(a, load_b_frag(&Bt[0][c * 16], 64, lane), acc[c]);
    __syncthreads();
  }

  const int mh = (lane >> 4) * 8;
  const int nl = lane & 15;
#pragma unroll
  for (int c = 0; c < 4; ++c) {
#pragma unroll
    for (int i = 0; i < 8; ++i) {
      int m = w * 16 + mh + i;
      int n = c * 16 + nl;
      int R = row0 + m, Cg = col0 + n;
      float val = (acc[c][i] + bias[Cg]) * scale;
      if (head_layout) {
        int b = R >> 10, s = R & 1023;            // S = 1024
        int h = Cg >> 6, d = Cg & 63;             // DK = 64
        Y[(((size_t)(b * H_NUM + h) * S_LEN + s) << 6) + d] = val;
      } else {
        Y[(size_t)R * N + Cg] = val;
      }
    }
  }
}

// ---------------------------------------------------------------------------
// K2: per-bh moments: mat = (qf^T qf + kf^T kf)/S + mu_q mu_k^T + mu_k mu_q^T
// ---------------------------------------------------------------------------
__global__ __launch_bounds__(256)
void moments_kernel(const float* __restrict__ q, const float* __restrict__ k,
                    float* __restrict__ mat) {
  __shared__ __align__(16) float Qs[64][64];
  __shared__ __align__(16) float Ks[64][64];
  __shared__ float muq[64], muk[64];
  const int bh = blockIdx.x;
  const int t = threadIdx.x;
  const float* qb = q + (size_t)bh * S_LEN * DKV;
  const float* kb = k + (size_t)bh * S_LEN * DKV;
  float acc1[16], acc2[16];
#pragma unroll
  for (int j = 0; j < 16; ++j) { acc1[j] = 0.f; acc2[j] = 0.f; }
  float mq = 0.f, mk = 0.f;

  const int sr = t >> 2;               // 0..63
  const int sc = (t & 3) << 4;         // 0,16,32,48

  for (int s0 = 0; s0 < S_LEN; s0 += 64) {
    {
      const float4* q4 = (const float4*)(qb + (size_t)(s0 + sr) * DKV + sc);
      const float4* k4 = (const float4*)(kb + (size_t)(s0 + sr) * DKV + sc);
#pragma unroll
      for (int u = 0; u < 4; ++u) {
        *(float4*)&Qs[sr][sc + 4 * u] = q4[u];
        *(float4*)&Ks[sr][sc + 4 * u] = k4[u];
      }
    }
    __syncthreads();
    if (t < 64) {
#pragma clang loop unroll_count(8)
      for (int r = 0; r < 64; ++r) { mq += Qs[r][t]; mk += Ks[r][t]; }
    }
#pragma unroll
    for (int j = 0; j < 16; ++j) {
      int idx = t + (j << 8);
      int d = idx >> 6, e = idx & 63;
      float a1 = acc1[j], a2 = acc2[j];
#pragma clang loop unroll_count(8)
      for (int r = 0; r < 64; ++r) {
        a1 += Qs[r][d] * Qs[r][e];
        a2 += Ks[r][d] * Ks[r][e];
      }
      acc1[j] = a1; acc2[j] = a2;
    }
    __syncthreads();
  }
  if (t < 64) { muq[t] = mq * (1.f / S_LEN); muk[t] = mk * (1.f / S_LEN); }
  __syncthreads();
#pragma unroll
  for (int j = 0; j < 16; ++j) {
    int idx = t + (j << 8);
    int d = idx >> 6, e = idx & 63;
    mat[(size_t)bh * 4096 + idx] =
        (acc1[j] + acc2[j]) * (1.f / S_LEN) + muq[d] * muk[e] + muk[d] * muq[e];
  }
}

// ---------------------------------------------------------------------------
// K3: per-bh symmetric eigendecomposition (cyclic Jacobi) of mat (64x64).
// SVD of a symmetric matrix: singular values |lambda| sorted descending,
// U columns = eigenvectors. Then Bmat = diag(sqrt(1-4a)) Q3^T, Dval=det^(1/4).
// ---------------------------------------------------------------------------
__global__ __launch_bounds__(64)
void jacobi_kernel(const float* __restrict__ mat, float* __restrict__ Bmat,
                   float* __restrict__ Dval) {
  __shared__ float A[64][64];
  __shared__ float V[64][64];
  __shared__ float cs[2];
  __shared__ float lam[64];
  __shared__ float om4a[64];
  __shared__ int perm[64];
  const int bh = blockIdx.x;
  const int t = threadIdx.x;

  for (int r = 0; r < 64; ++r) {
    A[r][t] = mat[(size_t)bh * 4096 + r * 64 + t];
    V[r][t] = (r == t) ? 1.f : 0.f;
  }
  __syncthreads();

#pragma clang loop unroll(disable)
  for (int sweep = 0; sweep < 8; ++sweep) {
#pragma clang loop unroll(disable)
    for (int p = 0; p < 63; ++p) {
#pragma clang loop unroll(disable)
      for (int q = p + 1; q < 64; ++q) {
        if (t == 0) {
          float app = A[p][p], aqq = A[q][q], apq = A[p][q];
          float c = 1.f, s = 0.f;
          if (fabsf(apq) > 1e-12f) {
            float tau = (aqq - app) / (2.f * apq);
            float sg = (tau >= 0.f) ? 1.f : -1.f;
            float tt = sg / (fabsf(tau) + sqrtf(1.f + tau * tau));
            c = rsqrtf(1.f + tt * tt);
            s = tt * c;
          }
          cs[0] = c; cs[1] = s;
        }
        __syncthreads();
        float c = cs[0], s = cs[1];
        float arp = A[p][t], arq = A[q][t];
        A[p][t] = c * arp - s * arq;
        A[q][t] = s * arp + c * arq;
        __syncthreads();
        float acp = A[t][p], acq = A[t][q];
        A[t][p] = c * acp - s * acq;
        A[t][q] = s * acp + c * acq;
        float vcp = V[t][p], vcq = V[t][q];
        V[t][p] = c * vcp - s * vcq;
        V[t][q] = s * vcp + c * vcq;
        __syncthreads();
      }
    }
  }

  lam[t] = fabsf(A[t][t]);
  __syncthreads();
  if (t == 0) {
    for (int i = 0; i < 64; ++i) perm[i] = i;
    for (int i = 0; i < 63; ++i) {              // sort descending (SVD order)
      int best = i;
      for (int j = i + 1; j < 64; ++j)
        if (lam[perm[j]] > lam[perm[best]]) best = j;
      int tmp = perm[i]; perm[i] = perm[best]; perm[best] = tmp;
    }
  }
  __syncthreads();

  const int src = perm[t];
  float l = lam[src];
  float a = (1.f - 2.f * l - sqrtf((2.f * l + 1.f) * (2.f * l + 1.f) + 8.f * l))
            * (1.f / 16.f);
  float o = 1.f - 4.f * a;                       // >= 1
  om4a[t] = o;
  float so = sqrtf(o);
  float* Brow = Bmat + (size_t)bh * 4096 + t * 64;
#pragma clang loop unroll_count(8)
  for (int d = 0; d < 64; ++d) Brow[d] = so * V[d][src];
  __syncthreads();
  if (t == 0) {
    float lp = 0.f;
    for (int e = 0; e < 64; ++e) lp += logf(om4a[e]);
    Dval[bh] = expf(0.25f * lp);                 // det(I-4A)^(1/4)
  }
}

// ---------------------------------------------------------------------------
// K4: omega = random_matrix @ omega_noise^T ; half_omega = 0.5*rowsum(omega^2)
// ---------------------------------------------------------------------------
__global__ __launch_bounds__(64)
void omega_kernel(const float* __restrict__ rm, const float* __restrict__ on,
                  float* __restrict__ half_omega) {
  __shared__ float RM[64][64];
  __shared__ float ON[64][64];
  const int t = threadIdx.x;
  for (int r = 0; r < 64; ++r) { RM[r][t] = rm[r * 64 + t]; ON[r][t] = on[r * 64 + t]; }
  __syncthreads();
  float acc = 0.f;
#pragma clang loop unroll(disable)
  for (int j = 0; j < 64; ++j) {
    float o = 0.f;
#pragma clang loop unroll_count(8)
    for (int u = 0; u < 64; ++u) o += RM[t][u] * ON[j][u];
    acc += o * o;
  }
  half_omega[t] = 0.5f * acc;
}

// ---------------------------------------------------------------------------
// K5: features (in place over qf/kf):
//   e = exp(qf @ Bmat_bh^T + half_omega[e] + Dval[e]); row-normalize (L2).
// WMMA: [64 rows x 64] x [64 x 64] per workgroup (4 waves x 16 rows).
// ---------------------------------------------------------------------------
__global__ __launch_bounds__(128)
void feature_kernel(float* __restrict__ qf, const float* __restrict__ Bmat,
                    const float* __restrict__ Dval,
                    const float* __restrict__ half_omega) {
  __shared__ __align__(16) _Float16 Bt[64][64];   // Bt[d][e] = Bmat[e][d]
  __shared__ __align__(16) _Float16 Qt[64][64];
  __shared__ float bias_e[64];
  const int bh = blockIdx.y;
  const int s0 = blockIdx.x * 64;
  const int t = threadIdx.x, lane = t & 31, w = t >> 5;

  const float* Bb = Bmat + (size_t)bh * 4096;
  float* qb = qf + ((size_t)bh * S_LEN + s0) * DKV;

  {
    const int e = t >> 1, ch = (t & 1) << 5;      // 32-float chunk
    // Bmat row -> transposed f16 tile
    const float4* b4 = (const float4*)(Bb + (size_t)e * 64 + ch);
    float4 g[8];
#pragma unroll
    for (int u = 0; u < 8; ++u) g[u] = b4[u];
#pragma unroll
    for (int u = 0; u < 8; ++u) {
      Bt[ch + 4 * u + 0][e] = (_Float16)g[u].x;
      Bt[ch + 4 * u + 1][e] = (_Float16)g[u].y;
      Bt[ch + 4 * u + 2][e] = (_Float16)g[u].z;
      Bt[ch + 4 * u + 3][e] = (_Float16)g[u].w;
    }
    // q rows -> row-major f16 tile (vectorized both sides)
    const float4* q4 = (const float4*)(qb + (size_t)e * DKV + ch);
#pragma unroll
    for (int u = 0; u < 4; ++u) {
      float4 f0 = q4[2 * u], f1 = q4[2 * u + 1];
      *(v8h*)&Qt[e][ch + 8 * u] = cvt_v8h(f0, f1);
    }
  }
  if (t < 64) bias_e[t] = half_omega[t] + Dval[t];
  __syncthreads();

  v16h a0 = load_a_frag(&Qt[w * 16][0], 64, lane);
  v16h a1 = load_a_frag(&Qt[w * 16][32], 64, lane);
  const v8f vzero = {0.f, 0.f, 0.f, 0.f, 0.f, 0.f, 0.f, 0.f};
  float ev[4][8];
  float ps[8];
#pragma unroll
  for (int i = 0; i < 8; ++i) ps[i] = 0.f;

#pragma unroll
  for (int c = 0; c < 4; ++c) {
    v8f acc = vzero;
    acc = wmma_f16(a0, load_b_frag(&Bt[0][c * 16], 64, lane), acc);
    acc = wmma_f16(a1, load_b_frag(&Bt[32][c * 16], 64, lane), acc);
    float be = bias_e[c * 16 + (lane & 15)];
#pragma unroll
    for (int i = 0; i < 8; ++i) {
      float vv = expf(acc[i] + be);
      ev[c][i] = vv;
      ps[i] += vv * vv;
    }
  }
  // row L2 norm: reduce across the 16-lane half holding this row's columns
#pragma unroll
  for (int msk = 8; msk >= 1; msk >>= 1)
#pragma unroll
    for (int i = 0; i < 8; ++i) ps[i] += __shfl_xor(ps[i], msk, 32);

  const int mh = (lane >> 4) * 8;
  const int nl = lane & 15;
#pragma unroll
  for (int i = 0; i < 8; ++i) {
    float inv = rsqrtf(ps[i]);
    int m = w * 16 + mh + i;
#pragma unroll
    for (int c = 0; c < 4; ++c)
      qb[(size_t)m * DKV + c * 16 + nl] = ev[c][i] * inv;
  }
}

// ---------------------------------------------------------------------------
// K6: flash attention per (bh, 64-row block); online softmax, 32-key tiles.
// scores: WMMA 16x16x32 (K=dk in 2 chunks); ctx: P(16x32) x V(32x64) WMMAs.
// ---------------------------------------------------------------------------
__global__ __launch_bounds__(128)
void flash_kernel(const float* __restrict__ qn, const float* __restrict__ kn,
                  const float* __restrict__ v, float* __restrict__ ctx) {
  __shared__ __align__(16) _Float16 Qt[64][64];
  __shared__ __align__(16) _Float16 Kt[64][32];   // Kt[d][j] (B layout, K=d)
  __shared__ __align__(16) _Float16 Vt[32][64];   // Vt[j][d] (B layout, K=j)
  __shared__ __align__(16) _Float16 Pw[4][16][32];
  const int bh = blockIdx.y;
  const int s0 = blockIdx.x * 64;
  const int t = threadIdx.x, lane = t & 31, w = t >> 5;

  const float* qb = qn + ((size_t)bh * S_LEN + s0) * DKV;
  const float* kb = kn + (size_t)bh * S_LEN * DKV;
  const float* vb = v + (size_t)bh * S_LEN * DKV;

  {
    const int r = t >> 1, ch = (t & 1) << 5;
    const float4* q4 = (const float4*)(qb + (size_t)r * DKV + ch);
#pragma unroll
    for (int u = 0; u < 4; ++u) {
      float4 f0 = q4[2 * u], f1 = q4[2 * u + 1];
      *(v8h*)&Qt[r][ch + 8 * u] = cvt_v8h(f0, f1);
    }
  }
  __syncthreads();
  v16h aq0 = load_a_frag(&Qt[w * 16][0], 64, lane);
  v16h aq1 = load_a_frag(&Qt[w * 16][32], 64, lane);

  const v8f vzero = {0.f, 0.f, 0.f, 0.f, 0.f, 0.f, 0.f, 0.f};
  v8f O[4];
#pragma unroll
  for (int c = 0; c < 4; ++c) O[c] = vzero;
  float mrow[8], lrow[8];
#pragma unroll
  for (int i = 0; i < 8; ++i) { mrow[i] = -1e30f; lrow[i] = 0.f; }

  const int mh = (lane >> 4) * 8;
  const int nl = lane & 15;
  const int sj = t >> 2;               // staging key (0..31)
  const int sd = (t & 3) << 4;         // staging d-chunk (0,16,32,48)

#pragma clang loop unroll(disable)
  for (int kt0 = 0; kt0 < S_LEN; kt0 += 32) {
    {
      // K tile: vector loads, transposed scalar stores
      const float4* k4 = (const float4*)(kb + (size_t)(kt0 + sj) * DKV + sd);
      float4 f0 = k4[0], f1 = k4[1], f2 = k4[2], f3 = k4[3];
      float tmp[16] = {f0.x, f0.y, f0.z, f0.w, f1.x, f1.y, f1.z, f1.w,
                       f2.x, f2.y, f2.z, f2.w, f3.x, f3.y, f3.z, f3.w};
#pragma unroll
      for (int u = 0; u < 16; ++u) Kt[sd + u][sj] = (_Float16)tmp[u];
      // V tile: vector loads + vector stores (row-major)
      const float4* v4p = (const float4*)(vb + (size_t)(kt0 + sj) * DKV + sd);
      float4 h0 = v4p[0], h1 = v4p[1], h2 = v4p[2], h3 = v4p[3];
      *(v8h*)&Vt[sj][sd]     = cvt_v8h(h0, h1);
      *(v8h*)&Vt[sj][sd + 8] = cvt_v8h(h2, h3);
    }
    if (kt0 + 32 < S_LEN) {            // prefetch next key/value tiles
      __builtin_prefetch(kb + (size_t)(kt0 + 32 + sj) * DKV + sd, 0, 1);
      __builtin_prefetch(vb + (size_t)(kt0 + 32 + sj) * DKV + sd, 0, 1);
    }
    __syncthreads();

    v8f st[2];
#pragma unroll
    for (int nc = 0; nc < 2; ++nc) {
      v8f acc = vzero;
      acc = wmma_f16(aq0, load_b_frag(&Kt[0][nc * 16], 32, lane), acc);
      acc = wmma_f16(aq1, load_b_frag(&Kt[32][nc * 16], 32, lane), acc);
      st[nc] = acc;
    }

    // online softmax per row (rows live across a 16-lane half)
    float p0[8], p1[8];
#pragma unroll
    for (int i = 0; i < 8; ++i) {
      float vmax = fmaxf(st[0][i], st[1][i]);
#pragma unroll
      for (int msk = 8; msk >= 1; msk >>= 1)
        vmax = fmaxf(vmax, __shfl_xor(vmax, msk, 32));
      float mnew = fmaxf(mrow[i], vmax);
      float f = expf(mrow[i] - mnew);
      float e0 = expf(st[0][i] - mnew);
      float e1 = expf(st[1][i] - mnew);
      float rs = e0 + e1;
#pragma unroll
      for (int msk = 8; msk >= 1; msk >>= 1)
        rs += __shfl_xor(rs, msk, 32);
      lrow[i] = lrow[i] * f + rs;
      mrow[i] = mnew;
      p0[i] = e0; p1[i] = e1;
#pragma unroll
      for (int c = 0; c < 4; ++c) O[c][i] = O[c][i] * f;
    }

    // C-layout -> A-layout via per-wave LDS round trip
#pragma unroll
    for (int i = 0; i < 8; ++i) {
      Pw[w][mh + i][nl] = (_Float16)p0[i];
      Pw[w][mh + i][nl + 16] = (_Float16)p1[i];
    }
    __syncthreads();
    v16h ap = load_a_frag(&Pw[w][0][0], 32, lane);
#pragma unroll
    for (int c = 0; c < 4; ++c)
      O[c] = wmma_f16(ap, load_b_frag(&Vt[0][c * 16], 64, lane), O[c]);
    __syncthreads();
  }

  // epilogue: divide by l, write ctx as [B*S, E] row-major
  const int b = bh >> 4, h = bh & 15;
#pragma unroll
  for (int i = 0; i < 8; ++i) {
    float inv = 1.f / lrow[i];
    int srow = s0 + w * 16 + mh + i;
#pragma unroll
    for (int c = 0; c < 4; ++c) {
      int d = c * 16 + nl;
      ctx[((size_t)(b * S_LEN + srow)) * E_DIM + h * 64 + d] = O[c][i] * inv;
    }
  }
}

// ---------------------------------------------------------------------------
// Host-side orchestration
// ---------------------------------------------------------------------------
extern "C" void kernel_launch(void* const* d_in, const int* in_sizes, int n_in,
                              void* d_out, int out_size, void* d_ws, size_t ws_size,
                              hipStream_t stream) {
  const float* hidden = (const float*)d_in[0];
  const float* Wq = (const float*)d_in[1];
  const float* Wk = (const float*)d_in[2];
  const float* Wv = (const float*)d_in[3];
  const float* Wo = (const float*)d_in[4];
  const float* bq = (const float*)d_in[5];
  const float* bk = (const float*)d_in[6];
  const float* bv = (const float*)d_in[7];
  const float* bo = (const float*)d_in[8];
  const float* rm = (const float*)d_in[9];
  const float* on = (const float*)d_in[10];

  float* ws = (float*)d_ws;
  const size_t SZ_QKV = (size_t)BH_NUM * S_LEN * DKV;   // 4,194,304
  float* q   = ws;                      // also qn (in-place feature map)
  float* k   = ws + SZ_QKV;             // also kn
  float* vv  = ws + 2 * SZ_QKV;
  float* mat = ws + 3 * SZ_QKV;         // 64*64*64
  float* Bm  = mat + (size_t)BH_NUM * 4096;
  float* Dv  = Bm + (size_t)BH_NUM * 4096;
  float* ho  = Dv + 64;
  float* ctx = ho + 64;                 // [B*S, E]

  const dim3 gGemm(E_DIM / 64, (B_NUM * S_LEN) / 64);   // (16, 64)
  const float scale = 0.125f;                           // 1/sqrt(64)

  gemm_xwt_kernel<<<gGemm, 128, 0, stream>>>(hidden, Wq, bq, q, B_NUM * S_LEN,
                                             E_DIM, E_DIM, scale, 1);
  gemm_xwt_kernel<<<gGemm, 128, 0, stream>>>(hidden, Wk, bk, k, B_NUM * S_LEN,
                                             E_DIM, E_DIM, scale, 1);
  gemm_xwt_kernel<<<gGemm, 128, 0, stream>>>(hidden, Wv, bv, vv, B_NUM * S_LEN,
                                             E_DIM, E_DIM, 1.0f, 1);

  moments_kernel<<<BH_NUM, 256, 0, stream>>>(q, k, mat);
  jacobi_kernel<<<BH_NUM, 64, 0, stream>>>(mat, Bm, Dv);
  omega_kernel<<<1, 64, 0, stream>>>(rm, on, ho);

  const dim3 gFeat(S_LEN / 64, BH_NUM);                 // (16, 64)
  feature_kernel<<<gFeat, 128, 0, stream>>>(q, Bm, Dv, ho);
  feature_kernel<<<gFeat, 128, 0, stream>>>(k, Bm, Dv, ho);

  flash_kernel<<<gFeat, 128, 0, stream>>>(q, k, vv, ctx);

  gemm_xwt_kernel<<<gGemm, 128, 0, stream>>>(ctx, Wo, bo, (float*)d_out,
                                             B_NUM * S_LEN, E_DIM, E_DIM, 1.0f, 0);
}
